// AttentionBlock_53730040873226
// MI455X (gfx1250) — matrix operand order repaired and checked by reading, not compiled, not run on hardware
//
#include <hip/hip_runtime.h>
#include <math.h>

#define B_  4
#define C_  1024
#define D_  1024
#define H_  16
#define HD_ 64
#define FF_ 4096

typedef __attribute__((ext_vector_type(16))) _Float16 v16h;
typedef __attribute__((ext_vector_type(8)))  _Float16 v8h;
typedef __attribute__((ext_vector_type(8)))  float    v8f;

// ---------------- WMMA fragment helpers (CDNA5 16x16x32 f16, wave32) ----------------

// A fragment: 16 rows x 32 K, row-major source, row stride ld (halfs).
// Layout (ISA 7.12.2): lane<16 -> row=lane,    K {0..7} in V0-3, K {16..23} in V4-7
//                      lane>=16 -> row=lane-16, K {8..15} in V0-3, K {24..31} in V4-7
static __device__ inline v16h load_afrag(const _Float16* base, int ld) {
  const int lane = threadIdx.x & 31;
  const int row  = lane & 15;
  const int koff = (lane >> 4) << 3;           // 0 or 8
  const _Float16* p = base + (size_t)row * ld + koff;
  v8h lo = *(const v8h*)p;                     // K = koff .. koff+7
  v8h hi = *(const v8h*)(p + 16);              // K = 16+koff .. 16+koff+7
  v16h r;
#pragma unroll
  for (int i = 0; i < 8; ++i) { r[i] = lo[i]; r[i + 8] = hi[i]; }
  return r;
}

// B fragment: 32 K x 16 N, from N-major storage baseN[n*ld + k].
// Layout: lanes 0-15 hold K {0..7,16..23} of column n=lane,
//         lanes 16-31 hold K {8..15,24..31} of column n=lane-16.
static __device__ inline v16h load_bfrag(const _Float16* baseN, int ld) {
  const int lane = threadIdx.x & 31;
  const int n    = lane & 15;
  const int kb   = (lane >> 4) << 3;           // 0 or 8
  const _Float16* p = baseN + (size_t)n * ld + kb;
  v8h lo = *(const v8h*)p;
  v8h hi = *(const v8h*)(p + 16);
  v16h r;
#pragma unroll
  for (int i = 0; i < 8; ++i) { r[i] = lo[i]; r[i + 8] = hi[i]; }
  return r;
}

static __device__ inline v8f wmma_f16(v16h a, v16h b, v8f c) {
  return __builtin_amdgcn_wmma_f32_16x16x32_f16(false, a, false, b, (short)0, c,
                                                false, false);
}

// 16-lane-group reductions (rows of a C-tile live across 16 lanes of a half-wave)
static __device__ inline float hmax16(float v) {
  v = fmaxf(v, __shfl_xor(v, 1));
  v = fmaxf(v, __shfl_xor(v, 2));
  v = fmaxf(v, __shfl_xor(v, 4));
  v = fmaxf(v, __shfl_xor(v, 8));
  return v;
}
static __device__ inline float hsum16(float v) {
  v += __shfl_xor(v, 1);
  v += __shfl_xor(v, 2);
  v += __shfl_xor(v, 4);
  v += __shfl_xor(v, 8);
  return v;
}

// ---------------- utility kernels ----------------

__global__ void zero_kernel(float* __restrict__ p, int n) {
  int i = blockIdx.x * 256 + threadIdx.x;
  if (i < n) p[i] = 0.0f;
}

// dst[batch][c][r] = (f16) src[batch][r][c]
__global__ void transpose_kernel(const float* __restrict__ src, _Float16* __restrict__ dst,
                                 int R, int Cc) {
  size_t bofs = (size_t)blockIdx.y * R * Cc;
  size_t idx  = (size_t)blockIdx.x * 256 + threadIdx.x;
  if (idx >= (size_t)R * Cc) return;
  int r = (int)(idx / Cc), c = (int)(idx % Cc);
  dst[bofs + (size_t)c * R + r] = (_Float16)src[bofs + idx];
}

// LayerNorm over rows of length 1024, f32 in -> f16 out
__global__ void ln_kernel(const float* __restrict__ x, const float* __restrict__ g,
                          const float* __restrict__ beta, _Float16* __restrict__ out) {
  __shared__ float red[256];
  const int tid = threadIdx.x;
  const float* xr = x + (size_t)blockIdx.x * D_;
  float s = 0.f;
  for (int i = tid; i < D_; i += 256) s += xr[i];
  red[tid] = s; __syncthreads();
  for (int off = 128; off > 0; off >>= 1) {
    if (tid < off) red[tid] += red[tid + off];
    __syncthreads();
  }
  float mean = red[0] * (1.0f / D_);
  __syncthreads();
  float v = 0.f;
  for (int i = tid; i < D_; i += 256) { float d = xr[i] - mean; v += d * d; }
  red[tid] = v; __syncthreads();
  for (int off = 128; off > 0; off >>= 1) {
    if (tid < off) red[tid] += red[tid + off];
    __syncthreads();
  }
  float rstd = rsqrtf(red[0] * (1.0f / D_) + 1e-5f);
  _Float16* orow = out + (size_t)blockIdx.x * D_;
  for (int i = tid; i < D_; i += 256)
    orow[i] = (_Float16)((xr[i] - mean) * rstd * g[i] + beta[i]);
}

// ---------------- QKV projections ----------------
// grid (B*H, C/64), block 128 (4 waves). Wave computes 16 rows x 64 outs for q,k,v.
__global__ void qkv_kernel(const _Float16* __restrict__ xn,
                           const _Float16* __restrict__ Wqt, const _Float16* __restrict__ Wkt,
                           const _Float16* __restrict__ Wvt,
                           const float* __restrict__ bq, const float* __restrict__ bk,
                           const float* __restrict__ bv,
                           _Float16* __restrict__ qm, _Float16* __restrict__ km,
                           _Float16* __restrict__ vT) {
  const int tid = threadIdx.x, lane = tid & 31, wid = tid >> 5;
  const int bh = blockIdx.x;
  const int b = bh >> 4, h = bh & 15;
  const int q0 = blockIdx.y * 64 + wid * 16;
  const int col = lane & 15, rb = (lane >> 4) << 3;
  const _Float16* abase = xn + ((size_t)b * C_ + q0) * D_ + h * HD_;
  v16h af0 = load_afrag(abase, D_);
  v16h af1 = load_afrag(abase + 32, D_);
  const _Float16* Ws[3] = {Wqt, Wkt, Wvt};
  const float*    bs[3] = {bq, bk, bv};
#pragma unroll
  for (int w = 0; w < 3; ++w) {
    const _Float16* Wt = Ws[w] + (size_t)h * HD_ * HD_;   // [out][in] N-major
#pragma unroll
    for (int nt = 0; nt < 4; ++nt) {
      v8f acc = {};
      acc = wmma_f16(af0, load_bfrag(Wt + (size_t)(nt * 16) * HD_, HD_), acc);
      acc = wmma_f16(af1, load_bfrag(Wt + (size_t)(nt * 16) * HD_ + 32, HD_), acc);
#pragma unroll
      for (int j = 0; j < 8; ++j) {
        int n = nt * 16 + col;
        int r = q0 + rb + j;
        float v = acc[j] + bs[w][h * HD_ + n];
        if (w == 0)      qm[((size_t)bh * C_ + r) * HD_ + n] = (_Float16)v;
        else if (w == 1) km[((size_t)bh * C_ + r) * HD_ + n] = (_Float16)v;
        else             vT[((size_t)bh * HD_ + n) * C_ + r] = (_Float16)v;  // transposed
      }
    }
  }
}

// ---------------- fused attention ----------------
// grid B*(C/16) blocks, 256 threads (8 waves). Wave w handles heads w and w+8 for
// the block's 16 query rows. Two-pass online softmax entirely in registers; the
// probability A-fragments for the A*V WMMA go through a per-wave LDS staging tile.
__global__ void attn_kernel(const _Float16* __restrict__ qm, const _Float16* __restrict__ km,
                            const _Float16* __restrict__ vT, const float* __restrict__ x,
                            float* __restrict__ xbp, float* __restrict__ attns) {
  __shared__ _Float16 astage[8][16 * 32];
  const int tid = threadIdx.x, lane = tid & 31, wid = tid >> 5;
  const int bid = blockIdx.x;
  const int b = bid >> 6;
  const int q0 = (bid & 63) << 4;
  const int col = lane & 15, rb = (lane >> 4) << 3;

  for (int hh = 0; hh < 2; ++hh) {
    const int h = wid + hh * 8;
    const size_t bh = (size_t)b * H_ + h;
    const _Float16* qb = qm + (bh * C_ + q0) * HD_;
    v16h af0 = load_afrag(qb, HD_);
    v16h af1 = load_afrag(qb + 32, HD_);

    float m[8], l[8];
#pragma unroll
    for (int j = 0; j < 8; ++j) { m[j] = -1e30f; l[j] = 0.f; }

    // pass 1: running row max + sum-of-exp (scores recomputed in pass 2)
    for (int n0 = 0; n0 < C_; n0 += 16) {
      const _Float16* kb = km + (bh * C_ + n0) * HD_;
      v8f s = {};
      s = wmma_f16(af0, load_bfrag(kb, HD_), s);
      s = wmma_f16(af1, load_bfrag(kb + 32, HD_), s);
#pragma unroll
      for (int j = 0; j < 8; ++j) {
        float sv = s[j] * 0.125f;                       // / sqrt(64)
        float nm = fmaxf(m[j], hmax16(sv));
        float es = hsum16(expf(sv - nm));
        l[j] = l[j] * expf(m[j] - nm) + es;
        m[j] = nm;
      }
    }
    float invl[8];
#pragma unroll
    for (int j = 0; j < 8; ++j) invl[j] = 1.0f / l[j];

    // pass 2: probabilities -> attns (mean over heads) and O = P @ V
    v8f oacc[4];
#pragma unroll
    for (int dt = 0; dt < 4; ++dt) oacc[dt] = (v8f){};

    for (int n0 = 0; n0 < C_; n0 += 32) {
      const _Float16* kb0 = km + (bh * C_ + n0) * HD_;
      const _Float16* kb1 = kb0 + 16 * HD_;
      v8f s0 = {}, s1 = {};
      s0 = wmma_f16(af0, load_bfrag(kb0, HD_), s0);
      s0 = wmma_f16(af1, load_bfrag(kb0 + 32, HD_), s0);
      s1 = wmma_f16(af0, load_bfrag(kb1, HD_), s1);
      s1 = wmma_f16(af1, load_bfrag(kb1 + 32, HD_), s1);
#pragma unroll
      for (int t = 0; t < 2; ++t) {
        v8f s = t ? s1 : s0;
#pragma unroll
        for (int j = 0; j < 8; ++j) {
          float p = expf(s[j] * 0.125f - m[j]) * invl[j];
          atomicAdd(&attns[((size_t)b * C_ + q0 + rb + j) * C_ + n0 + t * 16 + col],
                    p * (1.0f / H_));
          astage[wid][(rb + j) * 32 + t * 16 + col] = (_Float16)p;   // C-layout -> LDS
        }
      }
      asm volatile("s_wait_dscnt 0" ::: "memory");       // wave-local LDS RAW
      v16h pa = load_afrag(&astage[wid][0], 32);         // re-read in A-layout
#pragma unroll
      for (int dt = 0; dt < 4; ++dt) {
        const _Float16* vb = vT + (bh * HD_ + dt * 16) * C_ + n0;  // v^T: contiguous keys
        oacc[dt] = wmma_f16(pa, load_bfrag(vb, C_), oacc[dt]);
      }
      asm volatile("s_wait_dscnt 0" ::: "memory");       // WAR before next store round
    }

    // epilogue: x_bp = o + x (each wave owns its head's 64-dim slice of these rows)
#pragma unroll
    for (int dt = 0; dt < 4; ++dt)
#pragma unroll
      for (int j = 0; j < 8; ++j) {
        size_t idx = ((size_t)b * C_ + q0 + rb + j) * D_ + h * HD_ + dt * 16 + col;
        xbp[idx] = oacc[dt][j] + x[idx];
      }
  }
}

// ---------------- MLP GEMMs ----------------
// C = A[f16, MxK row-major] x B (Bt[f16, NxK N-major]) + bias, EPI 0: GELU->f16,
// EPI 1: + residual -> f32. Block 128 thr = 4 waves (2x2), wave tile 64x64,
// block tile 128x128, operands streamed from L2 (entire problem is L2-resident).
template <int EPI>
__global__ void gemm_kernel(const _Float16* __restrict__ A, const _Float16* __restrict__ Bt,
                            const float* __restrict__ bias, const float* __restrict__ res,
                            float* __restrict__ outF, _Float16* __restrict__ outH,
                            int M, int N, int K) {
  (void)M;
  const int tid = threadIdx.x, lane = tid & 31, wid = tid >> 5;
  const int m0 = blockIdx.x * 128 + (wid >> 1) * 64;
  const int n0 = blockIdx.y * 128 + (wid & 1) * 64;
  v8f acc[4][4];
#pragma unroll
  for (int rt = 0; rt < 4; ++rt)
#pragma unroll
    for (int ct = 0; ct < 4; ++ct) acc[rt][ct] = (v8f){};

  for (int k0 = 0; k0 < K; k0 += 32) {
    v16h af[4], bf[4];
#pragma unroll
    for (int rt = 0; rt < 4; ++rt)
      af[rt] = load_afrag(A + (size_t)(m0 + rt * 16) * K + k0, K);
#pragma unroll
    for (int ct = 0; ct < 4; ++ct)
      bf[ct] = load_bfrag(Bt + (size_t)(n0 + ct * 16) * K + k0, K);
    __builtin_prefetch(A + (size_t)m0 * K + k0 + 32, 0, 0);
#pragma unroll
    for (int rt = 0; rt < 4; ++rt)
#pragma unroll
      for (int ct = 0; ct < 4; ++ct)
        acc[rt][ct] = wmma_f16(af[rt], bf[ct], acc[rt][ct]);
  }

  const int col = lane & 15, rb = (lane >> 4) << 3;
#pragma unroll
  for (int rt = 0; rt < 4; ++rt)
#pragma unroll
    for (int ct = 0; ct < 4; ++ct)
#pragma unroll
      for (int j = 0; j < 8; ++j) {
        int mm = m0 + rt * 16 + rb + j;
        int nn = n0 + ct * 16 + col;
        float v = acc[rt][ct][j] + bias[nn];
        if (EPI == 0) {
          v = 0.5f * v * (1.0f + erff(v * 0.70710678118654752f));   // exact GELU
          outH[(size_t)mm * N + nn] = (_Float16)v;
        } else {
          outF[(size_t)mm * N + nn] = v + res[(size_t)mm * N + nn];
        }
      }
}

// ---------------- host launcher ----------------
extern "C" void kernel_launch(void* const* d_in, const int* in_sizes, int n_in,
                              void* d_out, int out_size, void* d_ws, size_t ws_size,
                              hipStream_t stream) {
  (void)in_sizes; (void)n_in; (void)out_size; (void)ws_size;
  const float* x     = (const float*)d_in[0];
  const float* g1    = (const float*)d_in[1];
  const float* b1    = (const float*)d_in[2];
  const float* Wq    = (const float*)d_in[3];
  const float* bq    = (const float*)d_in[4];
  const float* Wk    = (const float*)d_in[5];
  const float* bk    = (const float*)d_in[6];
  const float* Wv    = (const float*)d_in[7];
  const float* bv    = (const float*)d_in[8];
  const float* g2    = (const float*)d_in[9];
  const float* b2    = (const float*)d_in[10];
  const float* W1    = (const float*)d_in[11];
  const float* bias1 = (const float*)d_in[12];
  const float* W2    = (const float*)d_in[13];
  const float* bias2 = (const float*)d_in[14];

  float* out   = (float*)d_out;                       // [B,C,D]
  float* attns = out + (size_t)B_ * C_ * D_;          // [B,C,C]

  char* w = (char*)d_ws;
  auto take = [&](size_t bytes) -> char* {
    char* p = w;
    w += (bytes + 255) & ~(size_t)255;
    return p;
  };
  _Float16* xn   = (_Float16*)take((size_t)B_ * C_ * D_ * 2);
  _Float16* Wqt  = (_Float16*)take((size_t)H_ * HD_ * HD_ * 2);
  _Float16* Wkt  = (_Float16*)take((size_t)H_ * HD_ * HD_ * 2);
  _Float16* Wvt  = (_Float16*)take((size_t)H_ * HD_ * HD_ * 2);
  _Float16* W1t  = (_Float16*)take((size_t)D_ * FF_ * 2);
  _Float16* W2t  = (_Float16*)take((size_t)FF_ * D_ * 2);
  _Float16* qmw  = (_Float16*)take((size_t)B_ * H_ * C_ * HD_ * 2);
  _Float16* kmw  = (_Float16*)take((size_t)B_ * H_ * C_ * HD_ * 2);
  _Float16* vTw  = (_Float16*)take((size_t)B_ * H_ * HD_ * C_ * 2);
  float*    xbp  = (float*)take((size_t)B_ * C_ * D_ * 4);
  _Float16* xn2  = (_Float16*)take((size_t)B_ * C_ * D_ * 2);
  _Float16* hbuf = (_Float16*)take((size_t)B_ * C_ * FF_ * 2);

  // 0) zero attns (accumulated atomically below)
  zero_kernel<<<dim3((B_ * C_ * C_ + 255) / 256), 256, 0, stream>>>(attns, B_ * C_ * C_);

  // 1) weight prep: f32 -> f16, N-major (transposed)
  transpose_kernel<<<dim3((HD_ * HD_ + 255) / 256, H_), 256, 0, stream>>>(Wq, Wqt, HD_, HD_);
  transpose_kernel<<<dim3((HD_ * HD_ + 255) / 256, H_), 256, 0, stream>>>(Wk, Wkt, HD_, HD_);
  transpose_kernel<<<dim3((HD_ * HD_ + 255) / 256, H_), 256, 0, stream>>>(Wv, Wvt, HD_, HD_);
  transpose_kernel<<<dim3((D_ * FF_ + 255) / 256, 1), 256, 0, stream>>>(W1, W1t, D_, FF_);
  transpose_kernel<<<dim3((FF_ * D_ + 255) / 256, 1), 256, 0, stream>>>(W2, W2t, FF_, D_);

  // 2) LN1
  ln_kernel<<<B_ * C_, 256, 0, stream>>>(x, g1, b1, xn);

  // 3) QKV projections (WMMA)
  qkv_kernel<<<dim3(B_ * H_, C_ / 64), 128, 0, stream>>>(xn, Wqt, Wkt, Wvt, bq, bk, bv,
                                                         qmw, kmw, vTw);

  // 4) fused attention (WMMA scores + online softmax + WMMA O) + residual
  attn_kernel<<<B_ * (C_ / 16), 256, 0, stream>>>(qmw, kmw, vTw, x, xbp, attns);

  // 5) LN2
  ln_kernel<<<B_ * C_, 256, 0, stream>>>(xbp, g2, b2, xn2);

  // 6) MLP up-proj + GELU
  gemm_kernel<0><<<dim3((B_ * C_) / 128, FF_ / 128), 128, 0, stream>>>(
      xn2, W1t, bias1, nullptr, nullptr, hbuf, B_ * C_, FF_, D_);

  // 7) MLP down-proj + bias + residual -> out
  gemm_kernel<1><<<dim3((B_ * C_) / 128, D_ / 128), 128, 0, stream>>>(
      hbuf, W2t, bias2, xbp, out, nullptr, B_ * C_, D_, FF_);
}
